// T3STID_70841190580665
// MI455X (gfx1250) — matrix operand order repaired and checked by reading, not compiled, not run on hardware
//
#include <hip/hip_runtime.h>

// Problem constants (from reference): B,K,N,D,E,EE,H,P = 512,64,1024,128,100000,1024,256,12
#define BB   512
#define KK   64
#define NN   1024
#define DD   128
#define EEE  1024
#define HH   256
#define PP   12
#define HD   384   // H + D

typedef __attribute__((ext_vector_type(16))) __bf16 v16bf;
typedef __attribute__((ext_vector_type(8)))  __bf16 v8bf;
typedef __attribute__((ext_vector_type(8)))  float  v8f;

union Frag {
  v16bf v;
  v8bf  h[2];
  __bf16 e[16];
};

__device__ __forceinline__ __bf16 f2bf(float f) {
  unsigned u = __builtin_bit_cast(unsigned, f);
  u = (u + 0x7fffu + ((u >> 16) & 1u)) >> 16;   // round-to-nearest-even
  unsigned short s = (unsigned short)u;
  return __builtin_bit_cast(__bf16, s);
}

// ---------------------------------------------------------------------------
// Kernel A: per-b GEMM  EV[64,256] = gather(event_table)[64,1024] @ W_t^T,
// masked max over the 64 event rows (+b_t), then bias[b,p] =
//   ev_emb . W_mm[p,:H] + b_mm[p] + b_ts[p]   -> d_ws
// Block = 256 threads = 8 waves; wave w owns M-tile (w&3) and N-half (w>>2).
// ---------------------------------------------------------------------------
__global__ __launch_bounds__(256) void event_fuse_kernel(
    const float* __restrict__ event_table,
    const float* __restrict__ W_t,
    const float* __restrict__ b_t,
    const float* __restrict__ W_mm,
    const float* __restrict__ b_mm,
    const float* __restrict__ b_ts,
    const int*   __restrict__ event_idx,
    const unsigned char* __restrict__ event_mask,
    float* __restrict__ bias_out)
{
  __shared__ __bf16 evl[KK * 32];   // gathered event rows, 64 x 32 (EE chunk)
  __shared__ __bf16 wl[HH * 32];    // W_t chunk, 256 x 32
  __shared__ float  part[4 * HH];   // per-M-tile column maxes
  __shared__ float  cm[HH];         // fused event embedding (post-max, +b_t)
  __shared__ float  red[256];

  const int b    = blockIdx.x;
  const int t    = threadIdx.x;
  const int w    = t >> 5;
  const int lane = t & 31;
  const int mt   = w & 3;        // M tile (rows 16*mt .. 16*mt+15 of K=64)
  const int nh   = w >> 2;       // N half (8 tiles each)
  const int nlan = lane & 15;
  const int half = lane >> 4;    // 0: lanes 0-15, 1: lanes 16-31
  const int kbase = half ? 8 : 0;
  const int koff  = half ? 16 : 0;

  v8f acc[8] = {};

  for (int ec = 0; ec < EEE; ec += 32) {
    __syncthreads();
    // stage gathered event rows: 64 rows x 8 float4
    #pragma unroll
    for (int i = 0; i < 2; ++i) {
      int q   = t + i * 256;            // 0..511
      int row = q >> 3;
      int c4  = q & 7;
      int idx = event_idx[b * KK + row];
      const float4 v = *reinterpret_cast<const float4*>(
          event_table + (size_t)idx * EEE + ec + c4 * 4);
      int o = row * 32 + c4 * 4;
      evl[o+0] = f2bf(v.x); evl[o+1] = f2bf(v.y);
      evl[o+2] = f2bf(v.z); evl[o+3] = f2bf(v.w);
    }
    // stage W_t chunk: 256 rows x 8 float4
    #pragma unroll
    for (int i = 0; i < 8; ++i) {
      int q   = t + i * 256;            // 0..2047
      int row = q >> 3;
      int c4  = q & 7;
      const float4 v = *reinterpret_cast<const float4*>(
          W_t + (size_t)row * EEE + ec + c4 * 4);
      int o = row * 32 + c4 * 4;
      wl[o+0] = f2bf(v.x); wl[o+1] = f2bf(v.y);
      wl[o+2] = f2bf(v.z); wl[o+3] = f2bf(v.w);
    }
    __syncthreads();

    // A fragment: 16x32 bf16, lane m = lane&15, K = (j<8?0:16)+kbase+(j&7)
    Frag a;
    const __bf16* ap = &evl[(mt * 16 + nlan) * 32 + kbase];
    a.h[0] = *(const v8bf*)(ap);
    a.h[1] = *(const v8bf*)(ap + 16);

    #pragma unroll
    for (int i = 0; i < 8; ++i) {
      // B fragment: 32x16 bf16, lane n = lane&15, K = (j<8?0:16)+kbase+(j&7)
      const int ht = (nh * 8 + i) * 16 + nlan;
      const __bf16* bp = &wl[ht * 32 + kbase];
      Frag bfr;
      bfr.h[0] = *(const v8bf*)(bp);
      bfr.h[1] = *(const v8bf*)(bp + 16);
      acc[i] = __builtin_amdgcn_wmma_f32_16x16x32_bf16(
          false, a.v, false, bfr.v, (short)0, acc[i], false, false);
    }
  }

  // masked max over event rows (C/D layout: VGPR r -> m = r + half*8)
  #pragma unroll
  for (int i = 0; i < 8; ++i) {
    float m = -INFINITY;
    #pragma unroll
    for (int r = 0; r < 8; ++r) {
      int krow = mt * 16 + r + half * 8;
      if (event_mask[b * KK + krow]) m = fmaxf(m, acc[i][r]);
    }
    m = fmaxf(m, __shfl_xor(m, 16, 32));   // combine the two half-lane M sets
    if (half == 0) {
      int ht = (nh * 8 + i) * 16 + nlan;
      part[mt * HH + ht] = m;
    }
  }
  __syncthreads();
  if (t < HH) {
    float m = fmaxf(fmaxf(part[t], part[HH + t]),
                    fmaxf(part[2 * HH + t], part[3 * HH + t]));
    cm[t] = m + b_t[t];                    // +b_t commutes with max over k
  }
  __syncthreads();

  // bias[b][p] = cm . W_mm[p,:H] + b_mm[p] + b_ts[p]
  for (int p = 0; p < PP; ++p) {
    red[t] = cm[t] * W_mm[(size_t)p * HD + t];
    __syncthreads();
    for (int s = 128; s > 0; s >>= 1) {
      if (t < s) red[t] += red[t + s];
      __syncthreads();
    }
    if (t == 0) bias_out[b * PP + p] = red[0] + b_mm[p] + b_ts[p];
    __syncthreads();
  }
}

// ---------------------------------------------------------------------------
// Kernel B: out[b,p,n] = 0.5 * ( (W_ts + W_mm[:,H:]) @ ts_emb[b] + bias[b,p] )
// Block = 256 threads = 8 waves; wave w streams 8 N-tiles of 16 columns,
// K = D = 128 as 4 bf16 WMMA chunks. B fragments come straight from
// half-wave-coalesced global loads (each ts_emb element read exactly once).
// ---------------------------------------------------------------------------
__global__ __launch_bounds__(256) void forecast_kernel(
    const float* __restrict__ ts_emb,
    const float* __restrict__ W_mm,
    const float* __restrict__ W_ts,
    const float* __restrict__ bias_in,
    float* __restrict__ out)
{
  __shared__ __bf16 wc[16 * DD];   // combined weight, padded to 16 x 128
  __shared__ float  biasl[16];

  const int b    = blockIdx.x;
  const int t    = threadIdx.x;
  const int w    = t >> 5;
  const int lane = t & 31;
  const int nlan = lane & 15;
  const int half = lane >> 4;
  const int kbase = half ? 8 : 0;
  const int koff  = half ? 16 : 0;

  for (int e = t; e < 16 * DD; e += 256) {
    int p = e >> 7, d = e & 127;
    float v = (p < PP) ? (W_ts[(size_t)p * DD + d] + W_mm[(size_t)p * HD + HH + d])
                       : 0.0f;
    wc[e] = f2bf(v);
  }
  if (t < 16) biasl[t] = (t < PP) ? bias_in[b * PP + t] : 0.0f;
  __syncthreads();

  const size_t tsb = (size_t)b * DD * NN;

  #pragma unroll 1
  for (int i = 0; i < 8; ++i) {
    const int ntbase = (w * 8 + i) * 16;
    const int n = ntbase + nlan;
    v8f acc = {};
    #pragma unroll
    for (int kc = 0; kc < DD; kc += 32) {
      Frag a;
      const __bf16* ap = &wc[nlan * DD + kc + kbase];
      a.h[0] = *(const v8bf*)(ap);
      a.h[1] = *(const v8bf*)(ap + 16);

      Frag bfr;
      const float* src = ts_emb + tsb + (size_t)(kc + kbase) * NN + n;
      #pragma unroll
      for (int j = 0; j < 16; ++j)
        bfr.e[j] = f2bf(src[(size_t)(j < 8 ? j : j + 8) * NN]);   // B[k=(j<8?0:16)+kbase+(j&7)][n], coalesced in n

      acc = __builtin_amdgcn_wmma_f32_16x16x32_bf16(
          false, a.v, false, bfr.v, (short)0, acc, false, false);
    }
    #pragma unroll
    for (int r = 0; r < 8; ++r) {
      int p = r + half * 8;
      if (p < PP)
        out[((size_t)b * PP + p) * NN + n] = 0.5f * (acc[r] + biasl[p]);
    }
  }
}

extern "C" void kernel_launch(void* const* d_in, const int* in_sizes, int n_in,
                              void* d_out, int out_size, void* d_ws, size_t ws_size,
                              hipStream_t stream) {
  const float* ts_emb      = (const float*)d_in[0];
  const float* event_table = (const float*)d_in[1];
  const float* W_t         = (const float*)d_in[2];
  const float* b_t         = (const float*)d_in[3];
  const float* W_mm        = (const float*)d_in[4];
  const float* b_mm        = (const float*)d_in[5];
  const float* W_ts        = (const float*)d_in[6];
  const float* b_ts        = (const float*)d_in[7];
  const int*   event_idx   = (const int*)d_in[8];
  const unsigned char* event_mask = (const unsigned char*)d_in[9];
  float* out     = (float*)d_out;
  float* bias_ws = (float*)d_ws;   // [B, P] fused event bias

  event_fuse_kernel<<<dim3(BB), dim3(256), 0, stream>>>(
      event_table, W_t, b_t, W_mm, b_mm, b_ts, event_idx, event_mask, bias_ws);
  forecast_kernel<<<dim3(BB), dim3(256), 0, stream>>>(
      ts_emb, W_mm, W_ts, bias_ws, out);
}
